// LSA_43404939494068
// MI455X (gfx1250) — compile-verified, matrix operand order checked by reading, not run on hardware
//
#include <hip/hip_runtime.h>
#include <hip/hip_fp16.h>

#define B_ 64
#define T_ 1024
#define A_ 512
#define F_ 32
#define KW_ 31
#define PAD_ 15

typedef __attribute__((ext_vector_type(16))) _Float16 v16h;
typedef __attribute__((ext_vector_type(8)))  _Float16 v8h;
typedef __attribute__((ext_vector_type(8)))  float    v8f;

// Hardware tanh on CDNA5 (v_tanh_f32, TRANS-class, co-executes with VALU).
__device__ __forceinline__ float fast_tanh(float x)
{
#if __has_builtin(__builtin_amdgcn_tanhf)
    return __builtin_amdgcn_tanhf(x);
#else
    return tanhf(x);
#endif
}

// ---------------------------------------------------------------------------
// Kernel 1: location conv  loc[B,2,T] (*) conv_w[F,2,K] -> conv_out[B,T,F] f16
// One block = one (b, 256-t tile). Inputs + weights staged in LDS.
// ---------------------------------------------------------------------------
__global__ void __launch_bounds__(256) k_conv(
    const float* __restrict__ cum, const float* __restrict__ att,
    const float* __restrict__ conv_w, _Float16* __restrict__ convout)
{
    __shared__ float s_cw[F_ * 2 * KW_];
    __shared__ float s_cum[256 + 2 * PAD_];
    __shared__ float s_att[256 + 2 * PAD_];

    const int b    = blockIdx.x / (T_ / 256);
    const int t0   = (blockIdx.x % (T_ / 256)) * 256;
    const int tid  = threadIdx.x;

    for (int i = tid; i < F_ * 2 * KW_; i += 256) s_cw[i] = conv_w[i];
    for (int i = tid; i < 256 + 2 * PAD_; i += 256) {
        int t = t0 + i - PAD_;
        float cv = 0.f, av = 0.f;
        if (t >= 0 && t < T_) { cv = cum[b * T_ + t]; av = att[b * T_ + t]; }
        s_cum[i] = cv; s_att[i] = av;
    }
    __syncthreads();

    float cv[KW_], av[KW_];
#pragma unroll
    for (int k = 0; k < KW_; ++k) { cv[k] = s_cum[tid + k]; av[k] = s_att[tid + k]; }

    _Float16* outp = convout + ((size_t)b * T_ + (t0 + tid)) * F_;
    for (int f = 0; f < F_; ++f) {
        const float* w0 = &s_cw[(f * 2 + 0) * KW_];
        const float* w1 = &s_cw[(f * 2 + 1) * KW_];
        float s = 0.f;
#pragma unroll
        for (int k = 0; k < KW_; ++k) s = fmaf(w0[k], cv[k], fmaf(w1[k], av[k], s));
        outp[f] = (_Float16)s;
    }
}

// ---------------------------------------------------------------------------
// Kernel 2a: L_w f32 [A,F] -> f16 [A,F]
// ---------------------------------------------------------------------------
__global__ void k_cvt_lw(const float* __restrict__ Lw, _Float16* __restrict__ Lwh)
{
    int i = blockIdx.x * 256 + threadIdx.x;
    if (i < A_ * F_) Lwh[i] = (_Float16)Lw[i];
}

// ---------------------------------------------------------------------------
// Kernel 2b: pqlb[b,a] = query[b,:] . W_w[a,:] + W_b[a] + L_b[a]   (tiny GEMM)
// ---------------------------------------------------------------------------
__global__ void k_pq(const float* __restrict__ q, const float* __restrict__ Ww,
                     const float* __restrict__ Wb, const float* __restrict__ Lb,
                     float* __restrict__ pqlb)
{
    int i = blockIdx.x * 256 + threadIdx.x;          // i = b*A + a
    int b = i / A_, a = i % A_;
    const float* qr = q + (size_t)b * A_;
    const float* wr = Ww + (size_t)a * A_;
    float s = 0.f;
    for (int k = 0; k < A_; ++k) s = fmaf(qr[k], wr[k], s);
    pqlb[i] = s + Wb[a] + Lb[a];
}

// ---------------------------------------------------------------------------
// Kernel 3: fused energy pass (the bandwidth-bound one).
// Per wave: one 16-row t-tile. ploc tile via v_wmma_f32_16x16x32_f16
// (K=32 == F, so a single WMMA per 16x16 tile), then + pq + enc, v_tanh_f32,
// dot with v_w, cross-lane reduce -> u[b,t]. enc streamed once (134 MB, NT).
// ---------------------------------------------------------------------------
__global__ void __launch_bounds__(128) k_energy(
    const _Float16* __restrict__ convout, const _Float16* __restrict__ Lwh,
    const float* __restrict__ pqlb, const float* __restrict__ vw,
    const float* __restrict__ enc, float* __restrict__ u)
{
    const int wave = threadIdx.x >> 5;
    const int lane = threadIdx.x & 31;
    const int b    = blockIdx.x >> 4;
    const int tgrp = blockIdx.x & 15;
    const int t0   = (tgrp * 4 + wave) * 16;
    const int hi   = lane >> 4;      // half-wave selector in WMMA K layout
    const int nl   = lane & 15;      // M (for A) / N (for B, C, D)

    // A-fragment: 16x32 f16 layout — lane L(+16) holds row M=L,
    // K = {0..7,16..23} (lo half-wave) / {8..15,24..31} (hi half-wave).
    const v8h* rowp = (const v8h*)(convout + ((size_t)b * T_ + (t0 + nl)) * F_);
    v8h ac0 = rowp[hi];
    v8h ac1 = rowp[2 + hi];
    v16h afrag;
#pragma unroll
    for (int e = 0; e < 8; ++e) { afrag[e] = ac0[e]; afrag[e + 8] = ac1[e]; }

    float uacc[8];
#pragma unroll
    for (int r = 0; r < 8; ++r) uacc[r] = 0.f;

    for (int a0 = 0; a0 < A_; a0 += 16) {
        // B-fragment: 32x16 f16, B[k][n] = L_w[a0+n][k]; mirrored K split.
        const v8h* brow = (const v8h*)(Lwh + (size_t)(a0 + nl) * F_);
        v8h bc0 = brow[hi];
        v8h bc1 = brow[2 + hi];
        v16h bfrag;
#pragma unroll
        for (int e = 0; e < 8; ++e) { bfrag[e] = bc0[e]; bfrag[e + 8] = bc1[e]; }

        v8f c = {};
        c = __builtin_amdgcn_wmma_f32_16x16x32_f16(
                false, afrag, false, bfrag, (short)0, c, false, false);

        const int   a  = a0 + nl;
        const float pv = pqlb[b * A_ + a];
        const float vv = vw[a];
#pragma unroll
        for (int r = 0; r < 8; ++r) {
            const int t = t0 + r + 8 * hi;            // D layout: M = r (+8 hi-half)
            const float ev =
                __builtin_nontemporal_load(&enc[((size_t)b * T_ + t) * A_ + a]);
            const float x  = c[r] + pv + ev;
            uacc[r] = fmaf(vv, fast_tanh(x), uacc[r]);
        }
    }

    // Reduce over a: sum across each 16-lane half (N dimension).
#pragma unroll
    for (int r = 0; r < 8; ++r) {
        float v = uacc[r];
        v += __shfl_xor(v, 1, 32);
        v += __shfl_xor(v, 2, 32);
        v += __shfl_xor(v, 4, 32);
        v += __shfl_xor(v, 8, 32);
        if (nl == 0) u[(size_t)b * T_ + t0 + r + 8 * hi] = v;
    }
}

// ---------------------------------------------------------------------------
// Kernel 4: sigmoid, normalize, mask, forward recursion, normalize.
// One workgroup per batch row; LDS tree reductions.
// ---------------------------------------------------------------------------
__global__ void __launch_bounds__(256) k_final(
    const float* __restrict__ u, const float* __restrict__ alpha,
    const long long* __restrict__ plen, float* __restrict__ out)
{
    __shared__ float sarr[T_];
    __shared__ float red[256];
    const int b   = blockIdx.x;
    const int tid = threadIdx.x;
    const long long pl = plen[b];

    float lsum = 0.f;
    for (int t = tid; t < T_; t += 256) {
        float s = 1.f / (1.f + __expf(-u[(size_t)b * T_ + t]));
        sarr[t] = s;
        lsum += s;
    }
    red[tid] = lsum;
    __syncthreads();
    for (int o = 128; o > 0; o >>= 1) { if (tid < o) red[tid] += red[tid + o]; __syncthreads(); }
    const float ssum = red[0];
    __syncthreads();

    float na[T_ / 256];
    float lsum2 = 0.f;
#pragma unroll
    for (int i = 0; i < T_ / 256; ++i) {
        const int t  = tid + i * 256;
        const float m  = (t < (int)pl) ? 1.f : 0.f;
        const float at = (sarr[t] / ssum) * m;
        const float a0 = alpha[(size_t)b * T_ + t];
        const float a1 = (t >= 1) ? alpha[(size_t)b * T_ + t - 1] : 0.f;
        const float a2 = (t >= 2) ? alpha[(size_t)b * T_ + t - 2] : 0.f;
        const float v  = (a1 + a2 + a0 + 1e-7f) * at;
        na[i] = v;
        lsum2 += v;
    }
    red[tid] = lsum2;
    __syncthreads();
    for (int o = 128; o > 0; o >>= 1) { if (tid < o) red[tid] += red[tid + o]; __syncthreads(); }
    const float inv = 1.f / red[0];
#pragma unroll
    for (int i = 0; i < T_ / 256; ++i) {
        const int t = tid + i * 256;
        out[(size_t)b * T_ + t] = na[i] * inv;
    }
}

// ---------------------------------------------------------------------------
extern "C" void kernel_launch(void* const* d_in, const int* in_sizes, int n_in,
                              void* d_out, int out_size, void* d_ws, size_t ws_size,
                              hipStream_t stream)
{
    const float* enc    = (const float*)d_in[0];
    const float* query  = (const float*)d_in[1];
    const float* cum    = (const float*)d_in[2];
    const float* att    = (const float*)d_in[3];
    const float* alpha  = (const float*)d_in[4];
    const float* conv_w = (const float*)d_in[5];
    const float* L_w    = (const float*)d_in[6];
    const float* L_b    = (const float*)d_in[7];
    const float* W_w    = (const float*)d_in[8];
    const float* W_b    = (const float*)d_in[9];
    const float* v_w    = (const float*)d_in[10];
    /* d_in[11] = t (unused by the math) */
    const long long* plen = (const long long*)d_in[12];
    float* out = (float*)d_out;

    char* ws = (char*)d_ws;
    _Float16* convout = (_Float16*)ws;                                   // 4 MiB
    size_t off = (size_t)B_ * T_ * F_ * sizeof(_Float16);
    _Float16* Lwh = (_Float16*)(ws + off);                               // 32 KiB
    off += (size_t)A_ * F_ * sizeof(_Float16);
    float* pqlb = (float*)(ws + off);                                    // 128 KiB
    off += (size_t)B_ * A_ * sizeof(float);
    float* u = (float*)(ws + off);                                       // 256 KiB

    k_conv  <<<B_ * (T_ / 256), 256, 0, stream>>>(cum, att, conv_w, convout);
    k_cvt_lw<<<(A_ * F_ + 255) / 256, 256, 0, stream>>>(L_w, Lwh);
    k_pq    <<<(B_ * A_) / 256, 256, 0, stream>>>(query, W_w, W_b, L_b, pqlb);
    k_energy<<<B_ * 16, 128, 0, stream>>>(convout, Lwh, pqlb, v_w, enc, u);
    k_final <<<B_, 256, 0, stream>>>(u, alpha, plen, out);
}
